// QuantMultiheadAttention_76132590289061
// MI455X (gfx1250) — 8-bit matrix operand order repaired and checked by reading, compiled, not run on hardware
//
#include <hip/hip_runtime.h>
#include <hip/hip_bf16.h>
#include <stdint.h>

typedef __attribute__((ext_vector_type(8)))  int          v8i;
typedef __attribute__((ext_vector_type(4)))  int          v4i;
typedef __attribute__((ext_vector_type(4)))  unsigned int u32x4;
typedef __attribute__((ext_vector_type(8)))  int          i32x8;
typedef __attribute__((ext_vector_type(4)))  int          i32x4;

#define T_DIM 1024
#define S_DIM 2048
#define B_DIM 2
#define E_DIM 1024
#define H_DIM 16
#define HD    64
#define NH    (B_DIM * H_DIM)

enum { SLOT_Q = 0, SLOT_K, SLOT_V, SLOT_WQ, SLOT_WK, SLOT_WV, SLOT_WO,
       SLOT_QP, SLOT_KP, SLOT_VP, SLOT_AW, SLOT_AO, NUM_SLOTS = 16 };

#if __has_builtin(__builtin_amdgcn_tensor_load_to_lds) && __has_builtin(__builtin_amdgcn_s_wait_tensorcnt)
#define HAVE_TDM 1
#endif

__device__ __forceinline__ float slot_scale(const unsigned* s, int i) {
  return fmaxf(__uint_as_float(s[i]), 1e-8f) * (1.0f / 127.0f);
}

__global__ void k_init(unsigned* slots) {
  if (threadIdx.x < NUM_SLOTS) slots[threadIdx.x] = 0u;
}

__global__ void k_absmax(const float* __restrict__ x, int n, unsigned* __restrict__ slot) {
  float m = 0.0f;
  for (int i = blockIdx.x * blockDim.x + threadIdx.x; i < n; i += gridDim.x * blockDim.x)
    m = fmaxf(m, fabsf(x[i]));
#pragma unroll
  for (int o = 16; o; o >>= 1) m = fmaxf(m, __shfl_xor(m, o, 32));
  if ((threadIdx.x & 31) == 0) atomicMax(slot, __float_as_uint(m));
}

__global__ void k_quant(const float* __restrict__ x, int n,
                        const unsigned* __restrict__ slots, int si,
                        int8_t* __restrict__ out) {
  int i = blockIdx.x * blockDim.x + threadIdx.x;
  if (i >= n) return;
  float s = slot_scale(slots, si);
  float q = fminf(fmaxf(rintf(x[i] / s), -128.0f), 127.0f);
  out[i] = (int8_t)q;
}

__global__ void k_head_quant(const float* __restrict__ x, int L,
                             const unsigned* __restrict__ slots, int si,
                             int8_t* __restrict__ out) {
  int i = blockIdx.x * blockDim.x + threadIdx.x;
  if (i >= L * B_DIM * E_DIM) return;
  int e = i & (E_DIM - 1);
  int b = (i >> 10) & 1;
  int t = i >> 11;
  float s = slot_scale(slots, si);
  float q = fminf(fmaxf(rintf(x[i] / s), -128.0f), 127.0f);
  int h = e >> 6, ei = e & 63;
  out[(((size_t)(b * H_DIM + h) * (size_t)L + (size_t)t) << 6) + ei] = (int8_t)q;
}

__global__ __launch_bounds__(256) void k_gemm_i8(
    const int8_t* __restrict__ A, const int8_t* __restrict__ W,
    const float* __restrict__ bias, float* __restrict__ C,
    const unsigned* __restrict__ slots, int sa, int sw, float extra) {
  __shared__ int8_t At[128 * 80];
  __shared__ int8_t Bt[64 * 80];
  const int tid = threadIdx.x;
  const int wid = tid >> 5, lane = tid & 31;
  const int wy = wid >> 1, wx = wid & 1;
  const int lm = lane & 15, lh = lane >> 4;
  const int m0 = blockIdx.y * 128, n0 = blockIdx.x * 64;
  const int r = tid >> 2, qq = tid & 3;
  v8i acc[2][2] = {};
  for (int k0 = 0; k0 < E_DIM; k0 += 64) {
    v4i a0 = *(const v4i*)(A + (size_t)(m0 + r) * E_DIM + k0 + qq * 16);
    v4i a1 = *(const v4i*)(A + (size_t)(m0 + 64 + r) * E_DIM + k0 + qq * 16);
    if (k0 + 64 < E_DIM)
      __builtin_prefetch(A + (size_t)(m0 + r) * E_DIM + k0 + 64 + qq * 16, 0, 1);
    *(v4i*)(At + r * 80 + qq * 16) = a0;
    *(v4i*)(At + (64 + r) * 80 + qq * 16) = a1;
#ifdef HAVE_TDM
    if (wid == 0) {
      uint64_t ga = (uint64_t)(W + (size_t)n0 * E_DIM + k0);
      u32x4 g0;
      i32x8 g1 = {};
      i32x4 g2 = {}, g3 = {};
      g0[0] = 1u;
      g0[1] = (unsigned)(size_t)(&Bt[0]);
      g0[2] = (unsigned)(ga & 0xffffffffu);
      g0[3] = ((unsigned)(ga >> 32) & 0x01ffffffu) | 0x80000000u;
      g1[0] = (1 << 20) | (3 << 22) | (3 << 25);
      g1[1] = (int)(((unsigned)E_DIM & 0xffffu) << 16);
      g1[2] = (int)((((unsigned)E_DIM >> 16) & 0xffffu) |
                    (((unsigned)E_DIM & 0xffffu) << 16));
      g1[3] = (int)((((unsigned)E_DIM >> 16) & 0xffffu) | (64u << 16));
      g1[4] = 64;
      g1[5] = E_DIM;
      g1[6] = 0;
      g1[7] = 0;
#if __clang_major__ >= 23
      i32x8 g4 = {};
      __builtin_amdgcn_tensor_load_to_lds(g0, g1, g2, g3, g4, 0);
#else
      __builtin_amdgcn_tensor_load_to_lds(g0, g1, g2, g3, 0);
#endif
      __builtin_amdgcn_s_wait_tensorcnt((short)0);
    }
#else
    *(v4i*)(Bt + r * 80 + qq * 16) =
        *(const v4i*)(W + (size_t)(n0 + r) * E_DIM + k0 + qq * 16);
#endif
    __syncthreads();
    union { v8i v; long long d[4]; } af[2];
    union { v8i v; v4i q[2]; } bf[2];
#pragma unroll
    for (int i = 0; i < 2; i++) {
      const int8_t* ap = At + (wy * 32 + i * 16 + lm) * 80 + lh * 16;
      af[i].d[0] = *(const long long*)(ap);
      af[i].d[1] = *(const long long*)(ap + 8);
      af[i].d[2] = *(const long long*)(ap + 32);
      af[i].d[3] = *(const long long*)(ap + 40);
      const int8_t* bp = Bt + (wx * 32 + i * 16 + lm) * 80 + lh * 16;
      bf[i].q[0] = *(const v4i*)(bp);
      bf[i].q[1] = *(const v4i*)(bp + 32);
    }
#pragma unroll
    for (int i = 0; i < 2; i++)
#pragma unroll
      for (int j = 0; j < 2; j++)
        acc[i][j] = __builtin_amdgcn_wmma_i32_16x16x64_iu8(
            true, af[i].v, true, bf[j].v, acc[i][j], false, false);
    __syncthreads();
  }
  const float bs = slot_scale(slots, sa) * slot_scale(slots, sw);
#pragma unroll
  for (int i = 0; i < 2; i++)
#pragma unroll
    for (int j = 0; j < 2; j++) {
      int col = n0 + wx * 32 + j * 16 + lm;
      float bq = rintf(bias[col] / bs);
#pragma unroll
      for (int rr = 0; rr < 8; rr++) {
        int row = m0 + wy * 32 + i * 16 + rr + 8 * lh;
        C[(size_t)row * E_DIM + col] = ((float)acc[i][j][rr] + bq) * bs * extra;
      }
    }
}

__global__ __launch_bounds__(256) void k_attn_stats(
    const int8_t* __restrict__ qh, const int8_t* __restrict__ kh,
    float* __restrict__ mbuf, float* __restrict__ lbuf,
    const unsigned* __restrict__ slots, unsigned* __restrict__ awslot) {
  const int n = blockIdx.y;
  const int wid = threadIdx.x >> 5, lane = threadIdx.x & 31;
  const int lm = lane & 15, lh = lane >> 4;
  const int t0 = blockIdx.x * 128 + wid * 16;
  const float sqk = slot_scale(slots, SLOT_QP) * slot_scale(slots, SLOT_KP);
  union { v8i v; long long d[4]; } a;
  const int8_t* ap = qh + ((size_t)n * T_DIM + t0 + lm) * HD + lh * 16;
  a.d[0] = *(const long long*)(ap);
  a.d[1] = *(const long long*)(ap + 8);
  a.d[2] = *(const long long*)(ap + 32);
  a.d[3] = *(const long long*)(ap + 40);
  float rm[8], rl[8];
#pragma unroll
  for (int i = 0; i < 8; i++) { rm[i] = -3.0e38f; rl[i] = 0.0f; }
  for (int st = 0; st < S_DIM; st += 16) {
    union { v8i v; v4i q[2]; } b;
    const int8_t* bp = kh + ((size_t)n * S_DIM + st + lm) * HD + lh * 16;
    b.q[0] = *(const v4i*)(bp);
    b.q[1] = *(const v4i*)(bp + 32);
    v8i c = {};
    c = __builtin_amdgcn_wmma_i32_16x16x64_iu8(true, a.v, true, b.v, c, false, false);
#pragma unroll
    for (int i = 0; i < 8; i++) {
      float s = (float)c[i] * sqk;
      float tmax = s;
      tmax = fmaxf(tmax, __shfl_xor(tmax, 1, 32));
      tmax = fmaxf(tmax, __shfl_xor(tmax, 2, 32));
      tmax = fmaxf(tmax, __shfl_xor(tmax, 4, 32));
      tmax = fmaxf(tmax, __shfl_xor(tmax, 8, 32));
      float mn = fmaxf(rm[i], tmax);
      float e = __expf(s - mn);
      e += __shfl_xor(e, 1, 32);
      e += __shfl_xor(e, 2, 32);
      e += __shfl_xor(e, 4, 32);
      e += __shfl_xor(e, 8, 32);
      rl[i] = rl[i] * __expf(rm[i] - mn) + e;
      rm[i] = mn;
    }
  }
  if (lm == 0) {
    float im = 0.0f;
#pragma unroll
    for (int i = 0; i < 8; i++) {
      int row = t0 + i + 8 * lh;
      mbuf[(size_t)n * T_DIM + row] = rm[i];
      lbuf[(size_t)n * T_DIM + row] = rl[i];
      im = fmaxf(im, 1.0f / rl[i]);
    }
    atomicMax(awslot, __float_as_uint(im));
  }
}

__global__ __launch_bounds__(256) void k_attn_av(
    const int8_t* __restrict__ qh, const int8_t* __restrict__ kh,
    const int8_t* __restrict__ vh, const float* __restrict__ mbuf,
    const float* __restrict__ lbuf, const unsigned* __restrict__ slots,
    int8_t* __restrict__ pbuf, float* __restrict__ aout) {
  __shared__ int8_t vT[64 * 80];
  __shared__ int8_t pstage[8][16 * 64];
  const int n = blockIdx.y;
  const int bb = n >> 4, hh = n & 15;
  const int wid = threadIdx.x >> 5, lane = threadIdx.x & 31;
  const int lm = lane & 15, lh = lane >> 4;
  const int t0 = blockIdx.x * 128 + wid * 16;
  const float sqk = slot_scale(slots, SLOT_QP) * slot_scale(slots, SLOT_KP);
  const float saw = slot_scale(slots, SLOT_AW);
  const float inv_saw = 1.0f / saw;
  union { v8i v; long long d[4]; } a;
  const int8_t* ap = qh + ((size_t)n * T_DIM + t0 + lm) * HD + lh * 16;
  a.d[0] = *(const long long*)(ap);
  a.d[1] = *(const long long*)(ap + 8);
  a.d[2] = *(const long long*)(ap + 32);
  a.d[3] = *(const long long*)(ap + 40);
  float rm[8], rli[8];
#pragma unroll
  for (int i = 0; i < 8; i++) {
    int row = t0 + i + 8 * lh;
    rm[i] = mbuf[(size_t)n * T_DIM + row];
    rli[i] = 1.0f / lbuf[(size_t)n * T_DIM + row];
  }
  v8i oacc[4] = {};
  int8_t* myp = &pstage[wid][0];
  const int vcs = threadIdx.x & 63;
  const int vce = (threadIdx.x >> 6) * 16;
  for (int c0 = 0; c0 < S_DIM; c0 += 64) {
    __syncthreads();
    {
      v4i d = *(const v4i*)(vh + ((size_t)n * S_DIM + c0 + vcs) * HD + vce);
      const int8_t* db = (const int8_t*)&d;
#pragma unroll
      for (int i = 0; i < 16; i++) vT[(vce + i) * 80 + vcs] = db[i];
    }
    __syncthreads();
#pragma unroll
    for (int st = 0; st < 4; st++) {
      int s0 = c0 + st * 16;
      union { v8i v; v4i q[2]; } b;
      const int8_t* bp = kh + ((size_t)n * S_DIM + s0 + lm) * HD + lh * 16;
      b.q[0] = *(const v4i*)(bp);
      b.q[1] = *(const v4i*)(bp + 32);
      v8i c = {};
      c = __builtin_amdgcn_wmma_i32_16x16x64_iu8(true, a.v, true, b.v, c, false, false);
#pragma unroll
      for (int i = 0; i < 8; i++) {
        float sval = (float)c[i] * sqk;
        float p = __expf(sval - rm[i]) * rli[i];
        int qi = (int)fminf(rintf(p * inv_saw), 127.0f);
        int row = i + 8 * lh;
        myp[row * 64 + st * 16 + lm] = (int8_t)qi;
        pbuf[((size_t)n * T_DIM + t0 + row) * S_DIM + s0 + lm] = (int8_t)qi;
      }
    }
    asm volatile("s_wait_dscnt 0" ::: "memory");
    union { v8i v; long long d[4]; } pa;
    const int8_t* pp = myp + lm * 64 + lh * 16;
    pa.d[0] = *(const long long*)(pp);
    pa.d[1] = *(const long long*)(pp + 8);
    pa.d[2] = *(const long long*)(pp + 32);
    pa.d[3] = *(const long long*)(pp + 40);
#pragma unroll
    for (int et = 0; et < 4; et++) {
      union { v8i v; v4i q[2]; } vb;
      const int8_t* vp = vT + (et * 16 + lm) * 80 + lh * 16;
      vb.q[0] = *(const v4i*)(vp);
      vb.q[1] = *(const v4i*)(vp + 32);
      oacc[et] = __builtin_amdgcn_wmma_i32_16x16x64_iu8(
          true, pa.v, true, vb.v, oacc[et], false, false);
    }
  }
  const float osc = saw * slot_scale(slots, SLOT_VP);
#pragma unroll
  for (int et = 0; et < 4; et++)
#pragma unroll
    for (int i = 0; i < 8; i++) {
      int row = t0 + i + 8 * lh;
      int e = hh * 64 + et * 16 + lm;
      aout[(size_t)row * (B_DIM * E_DIM) + (size_t)bb * E_DIM + e] =
          (float)oacc[et][i] * osc;
    }
}

__global__ void k_avg(const int8_t* __restrict__ pbuf,
                      const unsigned* __restrict__ slots,
                      float* __restrict__ avg) {
  int i = blockIdx.x * blockDim.x + threadIdx.x;
  if (i >= B_DIM * T_DIM * (S_DIM / 4)) return;
  int s4 = i & (S_DIM / 4 - 1);
  int t = (i >> 9) & (T_DIM - 1);
  int b = i >> 19;
  int a0 = 0, a1 = 0, a2 = 0, a3 = 0;
#pragma unroll
  for (int h = 0; h < H_DIM; h++) {
    const int8_t* p = pbuf + (((size_t)(b * H_DIM + h) * T_DIM + t) * S_DIM) + s4 * 4;
    int w = *(const int*)p;
    a0 += (int)(int8_t)(w & 0xff);
    a1 += (int)(int8_t)((w >> 8) & 0xff);
    a2 += (int)(int8_t)((w >> 16) & 0xff);
    a3 += (int)(int8_t)(w >> 24);
  }
  float sc = slot_scale(slots, SLOT_AW) * (1.0f / (float)H_DIM);
  float4 o = make_float4(a0 * sc, a1 * sc, a2 * sc, a3 * sc);
  *(float4*)(avg + ((size_t)b * T_DIM + t) * S_DIM + s4 * 4) = o;
}

extern "C" void kernel_launch(void* const* d_in, const int* in_sizes, int n_in,
                              void* d_out, int out_size, void* d_ws, size_t ws_size,
                              hipStream_t stream) {
  (void)in_sizes; (void)n_in; (void)out_size; (void)ws_size;
  const float* query = (const float*)d_in[0];
  const float* key   = (const float*)d_in[1];
  const float* value = (const float*)d_in[2];
  const float* wq = (const float*)d_in[3];
  const float* bq = (const float*)d_in[4];
  const float* wk = (const float*)d_in[5];
  const float* bk = (const float*)d_in[6];
  const float* wv = (const float*)d_in[7];
  const float* bv = (const float*)d_in[8];
  const float* wo = (const float*)d_in[9];
  const float* bo = (const float*)d_in[10];

  uint8_t* ws = (uint8_t*)d_ws;
  size_t cur = 0;
  auto alloc = [&](size_t bytes) {
    cur = (cur + 255) & ~(size_t)255;
    size_t o = cur;
    cur += bytes;
    return o;
  };
  const size_t nq = (size_t)T_DIM * B_DIM * E_DIM;
  const size_t nk = (size_t)S_DIM * B_DIM * E_DIM;
  const size_t nw = (size_t)E_DIM * E_DIM;
  unsigned* slots = (unsigned*)(ws + alloc(NUM_SLOTS * 4));
  float* mbuf = (float*)(ws + alloc((size_t)NH * T_DIM * 4));
  float* lbuf = (float*)(ws + alloc((size_t)NH * T_DIM * 4));
  int8_t* xq  = (int8_t*)(ws + alloc(nq));
  int8_t* xk  = (int8_t*)(ws + alloc(nk));
  int8_t* xv  = (int8_t*)(ws + alloc(nk));
  int8_t* wq8 = (int8_t*)(ws + alloc(nw));
  int8_t* wk8 = (int8_t*)(ws + alloc(nw));
  int8_t* wv8 = (int8_t*)(ws + alloc(nw));
  int8_t* wo8 = (int8_t*)(ws + alloc(nw));
  float* qproj = (float*)(ws + alloc(nq * 4));
  float* kproj = (float*)(ws + alloc(nk * 4));
  float* vproj = (float*)(ws + alloc(nk * 4));
  int8_t* qh8 = (int8_t*)(ws + alloc((size_t)NH * T_DIM * HD));
  int8_t* kh8 = (int8_t*)(ws + alloc((size_t)NH * S_DIM * HD));
  int8_t* vh8 = (int8_t*)(ws + alloc((size_t)NH * S_DIM * HD));
  int8_t* pb  = (int8_t*)(ws + alloc((size_t)NH * T_DIM * S_DIM));
  float* aout = (float*)(ws + alloc(nq * 4));
  int8_t* ao8 = (int8_t*)(ws + alloc(nq));

  float* out  = (float*)d_out;
  float* avgw = out + nq;

  k_init<<<1, 32, 0, stream>>>(slots);

  k_absmax<<<512, 256, 0, stream>>>(query, (int)nq, slots + SLOT_Q);
  k_absmax<<<512, 256, 0, stream>>>(key,   (int)nk, slots + SLOT_K);
  k_absmax<<<512, 256, 0, stream>>>(value, (int)nk, slots + SLOT_V);
  k_absmax<<<512, 256, 0, stream>>>(wq, (int)nw, slots + SLOT_WQ);
  k_absmax<<<512, 256, 0, stream>>>(wk, (int)nw, slots + SLOT_WK);
  k_absmax<<<512, 256, 0, stream>>>(wv, (int)nw, slots + SLOT_WV);
  k_absmax<<<512, 256, 0, stream>>>(wo, (int)nw, slots + SLOT_WO);

  k_quant<<<(int)((nq + 255) / 256), 256, 0, stream>>>(query, (int)nq, slots, SLOT_Q, xq);
  k_quant<<<(int)((nk + 255) / 256), 256, 0, stream>>>(key,   (int)nk, slots, SLOT_K, xk);
  k_quant<<<(int)((nk + 255) / 256), 256, 0, stream>>>(value, (int)nk, slots, SLOT_V, xv);
  k_quant<<<(int)((nw + 255) / 256), 256, 0, stream>>>(wq, (int)nw, slots, SLOT_WQ, wq8);
  k_quant<<<(int)((nw + 255) / 256), 256, 0, stream>>>(wk, (int)nw, slots, SLOT_WK, wk8);
  k_quant<<<(int)((nw + 255) / 256), 256, 0, stream>>>(wv, (int)nw, slots, SLOT_WV, wv8);
  k_quant<<<(int)((nw + 255) / 256), 256, 0, stream>>>(wo, (int)nw, slots, SLOT_WO, wo8);

  dim3 gq(E_DIM / 64, (T_DIM * B_DIM) / 128);
  dim3 gk(E_DIM / 64, (S_DIM * B_DIM) / 128);
  k_gemm_i8<<<gq, 256, 0, stream>>>(xq, wq8, bq, qproj, slots, SLOT_Q, SLOT_WQ, 0.125f);
  k_gemm_i8<<<gk, 256, 0, stream>>>(xk, wk8, bk, kproj, slots, SLOT_K, SLOT_WK, 1.0f);
  k_gemm_i8<<<gk, 256, 0, stream>>>(xv, wv8, bv, vproj, slots, SLOT_V, SLOT_WV, 1.0f);

  k_absmax<<<512, 256, 0, stream>>>(qproj, (int)nq, slots + SLOT_QP);
  k_absmax<<<512, 256, 0, stream>>>(kproj, (int)nk, slots + SLOT_KP);
  k_absmax<<<512, 256, 0, stream>>>(vproj, (int)nk, slots + SLOT_VP);

  k_head_quant<<<(int)((nq + 255) / 256), 256, 0, stream>>>(qproj, T_DIM, slots, SLOT_QP, qh8);
  k_head_quant<<<(int)((nk + 255) / 256), 256, 0, stream>>>(kproj, S_DIM, slots, SLOT_KP, kh8);
  k_head_quant<<<(int)((nk + 255) / 256), 256, 0, stream>>>(vproj, S_DIM, slots, SLOT_VP, vh8);

  dim3 ga(T_DIM / 128, NH);
  k_attn_stats<<<ga, 256, 0, stream>>>(qh8, kh8, mbuf, lbuf, slots, slots + SLOT_AW);
  k_attn_av<<<ga, 256, 0, stream>>>(qh8, kh8, vh8, mbuf, lbuf, slots, pb, aout);

  k_avg<<<(B_DIM * T_DIM * (S_DIM / 4) + 255) / 256, 256, 0, stream>>>(pb, slots, avgw);

  k_absmax<<<512, 256, 0, stream>>>(aout, (int)nq, slots + SLOT_AO);
  k_quant<<<(int)((nq + 255) / 256), 256, 0, stream>>>(aout, (int)nq, slots, SLOT_AO, ao8);
  k_gemm_i8<<<gq, 256, 0, stream>>>(ao8, wo8, bo, out, slots, SLOT_AO, SLOT_WO, 1.0f);
}
